// CAF_26749056320313
// MI455X (gfx1250) — compile-verified, matrix operand order checked by reading, not run on hardware
//
#include <hip/hip_runtime.h>

typedef __attribute__((ext_vector_type(16))) __bf16 v16bf;
typedef __attribute__((ext_vector_type(8)))  float  v8f;
typedef __attribute__((ext_vector_type(8)))  unsigned short us8;
typedef __attribute__((ext_vector_type(4)))  unsigned short us4;

__device__ __forceinline__ unsigned short f2bf(float f) {
  unsigned int u = __float_as_uint(f);
  u = (u + 0x7FFFu + ((u >> 16) & 1u)) >> 16;
  return (unsigned short)u;
}
__device__ __forceinline__ float bf2f(unsigned short h) {
  return __uint_as_float(((unsigned int)h) << 16);
}

// ---------------------------------------------------------------------------
// Generic implicit-GEMM 3x1x1 / pointwise conv, bf16 A/B, f32 accumulate.
// out[b, m, t, n] = sum_{kt, ci} wA[m, kt*Cin+ci] * x[b, ci, t*ts+kt-tp, n]
// Block = 128 threads (4 waves). Block tile M=32 x N=128, K-step 64.
// Each wave owns a 32(M) x 32(N) register tile -> 4 accumulators, and issues
// 8 v_wmma_f32_16x16x32_bf16 per LDS stage (A/B fragments reused 2x each).
// Staging is vectorized: A via b128 copies, B via b64 global loads.
// Optional fused per-channel sum / sum-of-squares for training-mode BN.
// ---------------------------------------------------------------------------
__global__ __launch_bounds__(128)
void conv_wmma_kernel(float* __restrict__ out,
                      const unsigned short* __restrict__ xbf,
                      const unsigned short* __restrict__ wA,
                      const float* __restrict__ bias,
                      float* __restrict__ stat,
                      int Cout, int Cin, int KT, int Ktot,
                      int Tin, int Tout, int tstride, int tpad, int HW,
                      long long xbstride, long long xcstride, int awbstride)
{
  __shared__ unsigned short As[32 * 64];    // [m][k]   4 KB
  __shared__ unsigned short Bs[128 * 64];   // [n][k]  16 KB
  const int wave = threadIdx.x >> 5;
  const int lane = threadIdx.x & 31;
  const int half = lane >> 4;
  const int mr   = lane & 15;
  const int bt = blockIdx.y;
  const int b  = bt / Tout;
  const int t  = bt % Tout;
  const int mtile = blockIdx.z;             // 32-row tile
  const int n0b = blockIdx.x * 128;
  const unsigned short* Abase = wA + (long long)awbstride * b + (long long)mtile * 32 * Ktot;
  const unsigned short* xb = xbf + (long long)b * xbstride;

  const v8f vzero = {0.f, 0.f, 0.f, 0.f, 0.f, 0.f, 0.f, 0.f};
  v8f acc[2][2];
  acc[0][0] = vzero; acc[0][1] = vzero; acc[1][0] = vzero; acc[1][1] = vzero;

  const int kB   = threadIdx.x >> 1;        // 0..63: k row this thread stages
  const int nseg = (threadIdx.x & 1) * 64;  // 64-col segment

  for (int k0 = 0; k0 < Ktot; k0 += 64) {
    // stage A panel (32x64 bf16) with b128 vector copies (k0, rows 8-elem aligned)
    #pragma unroll
    for (int i = threadIdx.x * 8; i < 32 * 64; i += 128 * 8) {
      int m = i >> 6, k = i & 63;
      *(us8*)&As[i] = *(const us8*)&Abase[(long long)m * Ktot + (k0 + k)];
    }
    if (k0 + 64 < Ktot) __builtin_prefetch(&Abase[k0 + 64], 0, 1);
    // stage B panel (64k x 128n bf16): b64 global loads, transposed to LDS [n][k]
    {
      int kk = k0 + kB;
      int ci = kk % Cin;
      int kt = kk / Cin;
      int tt = t * tstride + kt - tpad;
      const unsigned short* src = nullptr;
      if (tt >= 0 && tt < Tin)
        src = xb + (long long)ci * xcstride + (long long)tt * HW;
      for (int j = 0; j < 64; j += 4) {
        int n = n0b + nseg + j;
        union { us4 v; unsigned short e[4]; } ld;
        ld.v = (us4){0, 0, 0, 0};
        if (src != nullptr && n < HW) ld.v = *(const us4*)&src[n];   // HW % 4 == 0
        #pragma unroll
        for (int qq = 0; qq < 4; ++qq)
          Bs[(nseg + j + qq) * 64 + kB] = ld.e[qq];
      }
    }
    __syncthreads();
    // 2 K-substeps x (2 A-frags x 2 B-frags) = 8 WMMAs per stage
    #pragma unroll
    for (int s = 0; s < 2; ++s) {
      union { v16bf v; unsigned int u[8]; } fa0, fa1, fb0, fb1;
      #pragma unroll
      for (int r = 0; r < 8; ++r) {
        int kb2 = 32 * s + ((r < 4) ? (r * 2 + half * 8) : (16 + (r - 4) * 2 + half * 8));
        fa0.u[r] = *(const unsigned int*)&As[mr * 64 + kb2];
        fa1.u[r] = *(const unsigned int*)&As[(16 + mr) * 64 + kb2];
        fb0.u[r] = *(const unsigned int*)&Bs[(wave * 32 + mr) * 64 + kb2];
        fb1.u[r] = *(const unsigned int*)&Bs[(wave * 32 + 16 + mr) * 64 + kb2];
      }
      acc[0][0] = __builtin_amdgcn_wmma_f32_16x16x32_bf16(false, fa0.v, false, fb0.v, (short)0, acc[0][0], false, false);
      acc[0][1] = __builtin_amdgcn_wmma_f32_16x16x32_bf16(false, fa0.v, false, fb1.v, (short)0, acc[0][1], false, false);
      acc[1][0] = __builtin_amdgcn_wmma_f32_16x16x32_bf16(false, fa1.v, false, fb0.v, (short)0, acc[1][0], false, false);
      acc[1][1] = __builtin_amdgcn_wmma_f32_16x16x32_bf16(false, fa1.v, false, fb1.v, (short)0, acc[1][1], false, false);
    }
    __syncthreads();
  }

  #pragma unroll
  for (int mt = 0; mt < 2; ++mt) {
    #pragma unroll
    for (int c = 0; c < 2; ++c) {
      int nbase = n0b + wave * 32 + c * 16;
      bool wact = nbase < HW;
      int n = nbase + mr;
      bool ncol = wact && (n < HW);
      #pragma unroll
      for (int r = 0; r < 8; ++r) {
        int m = mtile * 32 + mt * 16 + r + 8 * half;  // C/D layout: lanes>=16 -> M+8
        float val = acc[mt][c][r];
        if (bias != nullptr && m < Cout) val += bias[m];
        bool ok = ncol && (m < Cout);
        if (ok) out[(((long long)b * Cout + m) * Tout + t) * HW + n] = val;
        if (stat != nullptr) {
          float s  = ok ? val : 0.f;
          float s2 = s * s;
          #pragma unroll
          for (int d = 8; d >= 1; d >>= 1) {
            s  += __shfl_xor(s,  d, 32);
            s2 += __shfl_xor(s2, d, 32);
          }
          if (mr == 0 && m < Cout && wact) {
            atomicAdd(&stat[m], s);
            atomicAdd(&stat[Cout + m], s2);
          }
        }
      }
    }
  }
}

// ---------------------------------------------------------------------------
// BN (training-mode batch stats) + ReLU. Writes f32 and/or bf16 (into a
// channel-concat buffer with channel offset bfC0 / total channels bfCtot).
// ---------------------------------------------------------------------------
__global__ void bn_relu_kernel(const float* __restrict__ raw,
                               const float* __restrict__ stat,
                               const float* __restrict__ gam,
                               const float* __restrict__ bet,
                               int Cout, long long Nsp, int B, float invcnt,
                               float* __restrict__ f32dst,
                               unsigned short* __restrict__ bf1,
                               unsigned short* __restrict__ bf2,
                               int bfCtot, int bfC0)
{
  long long idx = (long long)blockIdx.x * blockDim.x + threadIdx.x;
  long long total = (long long)B * Cout * Nsp;
  if (idx >= total) return;
  long long rest = idx % Nsp;
  long long bc = idx / Nsp;
  int c = (int)(bc % Cout);
  long long b = bc / Cout;
  float m = stat[c] * invcnt;
  float var = stat[Cout + c] * invcnt - m * m;
  float y = (raw[idx] - m) * rsqrtf(var + 1e-5f) * gam[c] + bet[c];
  y = fmaxf(y, 0.f);
  if (f32dst) f32dst[idx] = y;
  if (bf1) {
    long long o = (b * bfCtot + (bfC0 + c)) * Nsp + rest;
    unsigned short h = f2bf(y);
    bf1[o] = h;
    if (bf2) bf2[o] = h;
  }
}

// Pack f32 weights (Cout,Cin,KT) -> bf16 A panel [Mpad][kt*Cin+ci], zero-padded rows.
__global__ void pack_w_kernel(const float* __restrict__ w, unsigned short* __restrict__ A,
                              int Cout, int Cin, int KT, int Mpad)
{
  long long Ktot = (long long)Cin * KT;
  long long idx = (long long)blockIdx.x * blockDim.x + threadIdx.x;
  if (idx >= (long long)Mpad * Ktot) return;
  int k = (int)(idx % Ktot);
  int m = (int)(idx / Ktot);
  int ci = k % Cin, kt = k / Cin;
  float v = (m < Cout) ? w[((long long)m * Cin + ci) * KT + kt] : 0.f;
  A[idx] = f2bf(v);
}

// f32 (B,C,T,HW) -> bf16 (B,C,T2,HW) at temporal offset t0 (builds concat(x1,x2)).
__global__ void cvt_in_kernel(const float* __restrict__ x, unsigned short* __restrict__ dst,
                              int T, int T2, int t0, int HW, long long total)
{
  long long idx = (long long)blockIdx.x * blockDim.x + threadIdx.x;
  if (idx >= total) return;
  int hw = (int)(idx % HW);
  long long tmp = idx / HW;
  int t = (int)(tmp % T);
  long long bc = tmp / T;
  dst[(bc * T2 + (t0 + t)) * HW + hw] = f2bf(x[idx]);
}

// dst = bf16(scale*a + x)
__global__ void axpy_bf_kernel(const float* __restrict__ a, const float* __restrict__ x,
                               const float* __restrict__ scale,
                               unsigned short* __restrict__ dst, long long total)
{
  long long idx = (long long)blockIdx.x * blockDim.x + threadIdx.x;
  if (idx >= total) return;
  dst[idx] = f2bf(scale[0] * a[idx] + x[idx]);
}

// ---------------------------------------------------------------------------
// PAM encoder: temporal mean + adaptive pools {1,2,3,6} -> p (B,64,50)
// ---------------------------------------------------------------------------
__global__ __launch_bounds__(256)
void pam_pool_kernel(const float* __restrict__ xi, float* __restrict__ p, int T, int HW)
{
  __shared__ float bins[50];
  int b = blockIdx.x >> 6;
  int c = blockIdx.x & 63;
  for (int i = threadIdx.x; i < 50; i += 256) bins[i] = 0.f;
  __syncthreads();
  const float* src = xi + ((long long)(b * 64 + c)) * T * HW;
  float invT = 1.f / (float)T;
  for (int hw = threadIdx.x; hw < HW; hw += 256) {
    float s = 0.f;
    for (int t = 0; t < T; ++t) s += src[(long long)t * HW + hw];
    s *= invT;
    int h = hw / 36, w = hw % 36;
    atomicAdd(&bins[0], s);
    atomicAdd(&bins[1 + (h / 18) * 2 + (w / 18)], s);
    atomicAdd(&bins[5 + (h / 12) * 3 + (w / 12)], s);
    atomicAdd(&bins[14 + (h / 6) * 6 + (w / 6)], s);
  }
  __syncthreads();
  for (int i = threadIdx.x; i < 50; i += 256) {
    float area = (i == 0) ? 1296.f : (i < 5) ? 324.f : (i < 14) ? 144.f : 36.f;
    p[((long long)(b * 64 + c)) * 50 + i] = bins[i] / area;
  }
}

// PAM per-scale 64x64 projection + BN(over B*s*s) + ReLU -> yenc (B,50,64)
__global__ __launch_bounds__(256)
void pam_proj_kernel(const float* __restrict__ p, const float* __restrict__ peW,
                     const float* __restrict__ peG, const float* __restrict__ peB,
                     float* __restrict__ yenc, int B)
{
  const int off[5] = {0, 1, 5, 14, 50};
  int s_id = blockIdx.x;
  int k0 = off[s_id], cnt = off[s_id + 1] - k0;
  int tot = B * cnt;                      // <= 144
  __shared__ float f[144 * 64];
  __shared__ float ms[64], vs[64];
  for (int it = threadIdx.x; it < tot * 64; it += 256) {
    int o = it & 63; int bp = it >> 6; int b = bp / cnt; int pos = bp % cnt;
    const float* wr = peW + ((long long)s_id * 64 + o) * 64;
    const float* pc = p + (long long)b * 64 * 50 + (k0 + pos);
    float acc = 0.f;
    for (int i = 0; i < 64; ++i) acc += wr[i] * pc[(long long)i * 50];
    f[it] = acc;
  }
  __syncthreads();
  for (int o = threadIdx.x; o < 64; o += 256) {
    float s = 0.f, s2 = 0.f;
    for (int j = 0; j < tot; ++j) { float v = f[j * 64 + o]; s += v; s2 += v * v; }
    float m = s / (float)tot;
    ms[o] = m; vs[o] = s2 / (float)tot - m * m;
  }
  __syncthreads();
  for (int it = threadIdx.x; it < tot * 64; it += 256) {
    int o = it & 63; int bp = it >> 6; int b = bp / cnt; int pos = bp % cnt;
    float y = (f[it] - ms[o]) * rsqrtf(vs[o] + 1e-5f) * peG[s_id * 64 + o] + peB[s_id * 64 + o];
    yenc[((long long)b * 50 + (k0 + pos)) * 64 + o] = fmaxf(y, 0.f);
  }
}

// PAM K/V projections: kbuf (B,50,16) f32; vpad (B,64,64) bf16 = V^T zero-padded.
__global__ void pam_kv_kernel(const float* __restrict__ yenc,
                              const float* __restrict__ wk, const float* __restrict__ bk,
                              const float* __restrict__ wv, const float* __restrict__ bv,
                              float* __restrict__ kbuf, unsigned short* __restrict__ vpad)
{
  int b = blockIdx.x;
  for (int it = threadIdx.x; it < 50 * 16; it += 256) {
    int kk = it / 16, o = it % 16;
    float a = bk[o];
    const float* yr = yenc + ((long long)b * 50 + kk) * 64;
    for (int i = 0; i < 64; ++i) a += yr[i] * wk[o * 64 + i];
    kbuf[((long long)b * 50 + kk) * 16 + o] = a;
  }
  for (int it = threadIdx.x; it < 64 * 64; it += 256) {
    int c = it / 64, kk = it % 64;
    float a = 0.f;
    if (kk < 50) {
      a = bv[c];
      const float* yr = yenc + ((long long)b * 50 + kk) * 64;
      for (int i = 0; i < 64; ++i) a += yr[i] * wv[c * 64 + i];
    }
    vpad[((long long)b * 64 + c) * 64 + kk] = f2bf(a);
  }
}

// PAM attention: softmax over K=50 of q.k, written k-major bf16, zero-padded to 64.
__global__ void pam_attn_kernel(const float* __restrict__ q, const float* __restrict__ kb,
                                unsigned short* __restrict__ attn, int Nsp, long long total)
{
  long long idx = (long long)blockIdx.x * blockDim.x + threadIdx.x;
  if (idx >= total) return;
  int n = (int)(idx % Nsp);
  int b = (int)(idx / Nsp);
  float qv[16];
  #pragma unroll
  for (int d = 0; d < 16; ++d) qv[d] = q[((long long)b * 16 + d) * Nsp + n];
  const float* kbase = kb + (long long)b * 50 * 16;
  float mx = -1e30f;
  for (int kk = 0; kk < 50; ++kk) {
    float a = 0.f;
    #pragma unroll
    for (int d = 0; d < 16; ++d) a += qv[d] * kbase[kk * 16 + d];
    mx = fmaxf(mx, a);
  }
  float s = 0.f;
  for (int kk = 0; kk < 50; ++kk) {
    float a = 0.f;
    #pragma unroll
    for (int d = 0; d < 16; ++d) a += qv[d] * kbase[kk * 16 + d];
    s += __expf(a - mx);
  }
  float inv = 1.f / s;
  unsigned short* orow = attn + (long long)b * 64 * Nsp + n;
  for (int kk = 0; kk < 50; ++kk) {
    float a = 0.f;
    #pragma unroll
    for (int d = 0; d < 16; ++d) a += qv[d] * kbase[kk * 16 + d];
    orow[(long long)kk * Nsp] = f2bf(__expf(a - mx) * inv);
  }
  for (int kk = 50; kk < 64; ++kk) orow[(long long)kk * Nsp] = 0;
}

// CAM energy: energy[b,c,k] = sum_n x[c,n]*y[k,n], k<4.
__global__ __launch_bounds__(256)
void cam_energy_kernel(const float* __restrict__ xi, const float* __restrict__ yc,
                       float* __restrict__ energy, int Nsp)
{
  int b = blockIdx.x >> 6;
  int c = blockIdx.x & 63;
  const float* xr = xi + ((long long)(b * 64 + c)) * Nsp;
  const float* y0 = yc + (long long)b * 4 * Nsp;
  float a[4] = {0.f, 0.f, 0.f, 0.f};
  for (int n = threadIdx.x; n < Nsp; n += 256) {
    float xv = xr[n];
    #pragma unroll
    for (int k = 0; k < 4; ++k) a[k] += xv * y0[(long long)k * Nsp + n];
  }
  __shared__ float sm[4][8];
  int lane = threadIdx.x & 31, wv = threadIdx.x >> 5;
  #pragma unroll
  for (int k = 0; k < 4; ++k) {
    float v = a[k];
    #pragma unroll
    for (int d = 16; d >= 1; d >>= 1) v += __shfl_xor(v, d, 32);
    if (lane == 0) sm[k][wv] = v;
  }
  __syncthreads();
  if (threadIdx.x < 4) {
    float s = 0.f;
    #pragma unroll
    for (int w8 = 0; w8 < 8; ++w8) s += sm[threadIdx.x][w8];
    energy[(long long)blockIdx.x * 4 + threadIdx.x] = s;
  }
}

// CAM apply: attn = softmax_k(max-e) -> out = x + scale * attn . y
__global__ void cam_apply_kernel(const float* __restrict__ xi, const float* __restrict__ yc,
                                 const float* __restrict__ energy, const float* __restrict__ scale,
                                 unsigned short* __restrict__ dst, int Nsp, long long total)
{
  long long idx = (long long)blockIdx.x * blockDim.x + threadIdx.x;
  if (idx >= total) return;
  int n = (int)(idx % Nsp);
  long long bc = idx / Nsp;
  int c = (int)(bc & 63);
  int b = (int)(bc >> 6);
  const float* e = energy + ((long long)(b * 64 + c)) * 4;
  float mn = fminf(fminf(e[0], e[1]), fminf(e[2], e[3]));
  float w0 = __expf(mn - e[0]), w1 = __expf(mn - e[1]);
  float w2 = __expf(mn - e[2]), w3 = __expf(mn - e[3]);
  float s = w0 + w1 + w2 + w3;
  const float* y0 = yc + (long long)b * 4 * Nsp + n;
  float o = (w0 * y0[0] + w1 * y0[(long long)Nsp] + w2 * y0[2LL * Nsp] + w3 * y0[3LL * Nsp]) / s;
  dst[idx] = f2bf(xi[idx] + scale[0] * o);
}

// TAM energy: energy[b,i,j] = sum_{c,hw} x1[c,i,hw]*x2[c,j,hw]
__global__ __launch_bounds__(256)
void tam_energy_kernel(const float* __restrict__ x1f, const float* __restrict__ x2f,
                       float* __restrict__ energy, int T2, int HW)
{
  int i = blockIdx.x, j = blockIdx.y, b = blockIdx.z;
  const float* p = x1f + ((long long)b * 64 * T2 + i) * HW;
  const float* q = x2f + ((long long)b * 64 * T2 + j) * HW;
  float a = 0.f;
  int M = 64 * HW;
  for (int it = threadIdx.x; it < M; it += 256) {
    int c = it / HW, hw = it % HW;
    long long o = (long long)c * T2 * HW + hw;
    a += p[o] * q[o];
  }
  int lane = threadIdx.x & 31, wv = threadIdx.x >> 5;
  #pragma unroll
  for (int d = 16; d >= 1; d >>= 1) a += __shfl_xor(a, d, 32);
  __shared__ float sm[8];
  if (lane == 0) sm[wv] = a;
  __syncthreads();
  if (threadIdx.x == 0) {
    float s = 0.f;
    #pragma unroll
    for (int k = 0; k < 8; ++k) s += sm[k];
    energy[((long long)b * T2 + i) * T2 + j] = s;
  }
}

__global__ void tam_attn_kernel(const float* __restrict__ energy, float* __restrict__ attn, int T2)
{
  int b = blockIdx.x, i = threadIdx.x;                 // blockDim = T2
  const float* e = energy + ((long long)b * T2 + i) * T2;
  float mn = 1e30f;
  for (int j = 0; j < T2; ++j) mn = fminf(mn, e[j]);   // softmax(max - e) == exp(mn - e)/sum
  float s = 0.f;
  for (int j = 0; j < T2; ++j) s += __expf(mn - e[j]);
  float inv = 1.f / s;
  float* o = attn + ((long long)b * T2 + i) * T2;
  for (int j = 0; j < T2; ++j) o[j] = __expf(mn - e[j]) * inv;
}

__global__ void tam_apply_kernel(const float* __restrict__ x1f, const float* __restrict__ x2f,
                                 const float* __restrict__ attn, const float* __restrict__ scale,
                                 unsigned short* __restrict__ dst, int T2, int HW, long long total)
{
  long long idx = (long long)blockIdx.x * blockDim.x + threadIdx.x;
  if (idx >= total) return;
  int hw = (int)(idx % HW);
  long long r = idx / HW;
  int t = (int)(r % T2);
  long long bc = r / T2;                                // b*64+c
  const float* arow = attn + ((long long)(bc >> 6) * T2 + t) * T2;
  const float* q = x2f + bc * T2 * HW + hw;
  float a = 0.f;
  for (int j = 0; j < T2; ++j) a += arow[j] * q[(long long)j * HW];
  dst[idx] = f2bf(x1f[idx] + scale[0] * a);
}

// ---------------------------------------------------------------------------
extern "C" void kernel_launch(void* const* d_in, const int* in_sizes, int n_in,
                              void* d_out, int out_size, void* d_ws, size_t ws_size,
                              hipStream_t stream)
{
  (void)in_sizes; (void)n_in; (void)out_size; (void)ws_size;
  const int B = 4, C = 256, T = 16, T2 = 32, HW = 1296;
  const int Nsp = T * HW;                 // 20736
  const long long N2 = (long long)T2 * HW; // 41472

  const float* x1      = (const float*)d_in[0];
  const float* x2      = (const float*)d_in[1];
  const float* w_conv1 = (const float*)d_in[2];
  const float* g_conv1 = (const float*)d_in[3];
  const float* b_conv1 = (const float*)d_in[4];
  const float* w_conv2 = (const float*)d_in[5];
  const float* g_conv2 = (const float*)d_in[6];
  const float* b_conv2 = (const float*)d_in[7];
  const float* w_tam   = (const float*)d_in[8];
  const float* g_tam   = (const float*)d_in[9];
  const float* b_tam   = (const float*)d_in[10];
  const float* w_cat   = (const float*)d_in[11];
  const float* g_cat   = (const float*)d_in[12];
  const float* b_cat   = (const float*)d_in[13];
  const float* pe_w    = (const float*)d_in[14];
  const float* pe_g    = (const float*)d_in[15];
  const float* pe_b    = (const float*)d_in[16];
  const float* pd_wq   = (const float*)d_in[17];
  const float* pd_bq   = (const float*)d_in[18];
  const float* pd_wk   = (const float*)d_in[19];
  const float* pd_bk   = (const float*)d_in[20];
  const float* pd_wv   = (const float*)d_in[21];
  const float* pd_bv   = (const float*)d_in[22];
  const float* pd_scale= (const float*)d_in[23];
  const float* ce_w    = (const float*)d_in[24];
  const float* ce_g    = (const float*)d_in[25];
  const float* ce_b    = (const float*)d_in[26];
  const float* cd_scale= (const float*)d_in[27];
  const float* te_w    = (const float*)d_in[28];
  const float* te_g    = (const float*)d_in[29];
  const float* te_b    = (const float*)d_in[30];
  const float* td_scale= (const float*)d_in[31];

  char* wsb = (char*)d_ws;
  size_t off = 0;
  auto alloc = [&](size_t bytes) -> void* {
    void* p = wsb + off;
    off += (bytes + 255) & ~(size_t)255;
    return p;
  };

  unsigned short* xcat_bf = (unsigned short*)alloc((size_t)B * C * T2 * HW * 2);
  unsigned short* wA1  = (unsigned short*)alloc((size_t)64 * 768 * 2);
  unsigned short* wA2  = (unsigned short*)alloc((size_t)64 * 192 * 2);
  unsigned short* wAt  = (unsigned short*)alloc((size_t)64 * 192 * 2);
  unsigned short* wAc  = (unsigned short*)alloc((size_t)256 * 576 * 2);
  unsigned short* wAq  = (unsigned short*)alloc((size_t)32 * 64 * 2);
  unsigned short* wAte = (unsigned short*)alloc((size_t)64 * 64 * 2);
  unsigned short* wAce = (unsigned short*)alloc((size_t)32 * 64 * 2);
  float* R = (float*)alloc((size_t)B * 256 * Nsp * 4);            // pre-BN scratch (max tensor)
  float* x1i_f = (float*)alloc((size_t)B * 64 * Nsp * 4);
  unsigned short* x1i_b = (unsigned short*)alloc((size_t)B * 64 * Nsp * 2);
  float* x2i_f = (float*)alloc((size_t)B * 64 * Nsp * 4);
  unsigned short* x2i_b = (unsigned short*)alloc((size_t)B * 64 * Nsp * 2);
  float* xt1_f = (float*)alloc((size_t)B * 64 * N2 * 4);
  unsigned short* xt1_b = (unsigned short*)alloc((size_t)B * 64 * N2 * 2);
  float* xt2_f = (float*)alloc((size_t)B * 64 * N2 * 4);
  unsigned short* tam_b = (unsigned short*)alloc((size_t)B * 64 * N2 * 2);
  unsigned short* catA = (unsigned short*)alloc((size_t)B * 192 * Nsp * 2);
  unsigned short* catB = (unsigned short*)alloc((size_t)B * 192 * Nsp * 2);
  float* qbuf = (float*)alloc((size_t)B * 16 * Nsp * 4);
  unsigned short* attnP = (unsigned short*)alloc((size_t)B * 64 * Nsp * 2);
  float* pamout = (float*)alloc((size_t)B * 64 * Nsp * 4);
  unsigned short* pres_b = (unsigned short*)alloc((size_t)B * 64 * Nsp * 2);
  unsigned short* cres_b = (unsigned short*)alloc((size_t)B * 64 * Nsp * 2);
  float* yc    = (float*)alloc((size_t)B * 4 * Nsp * 4);
  float* ppool = (float*)alloc((size_t)B * 64 * 50 * 4);
  float* yenc  = (float*)alloc((size_t)B * 50 * 64 * 4);
  float* kbuf  = (float*)alloc((size_t)B * 50 * 16 * 4);
  unsigned short* vpad = (unsigned short*)alloc((size_t)B * 64 * 64 * 2);
  float* eCam  = (float*)alloc((size_t)B * 64 * 4 * 4);
  float* eTam  = (float*)alloc((size_t)B * 32 * 32 * 4);
  float* aTam  = (float*)alloc((size_t)B * 32 * 32 * 4);
  float* stats = (float*)alloc((size_t)16 * 512 * 4);

  hipMemsetAsync(stats, 0, 16 * 512 * 4, stream);

  auto ew = [](long long total) { return dim3((unsigned)((total + 255) / 256)); };

  // weight packing + input bf16 conversion (builds concat(x1,x2) along T)
  pack_w_kernel<<<ew(64LL * 768), 256, 0, stream>>>(w_conv1, wA1, 64, 256, 3, 64);
  pack_w_kernel<<<ew(64LL * 192), 256, 0, stream>>>(w_conv2, wA2, 64, 64, 3, 64);
  pack_w_kernel<<<ew(64LL * 192), 256, 0, stream>>>(w_tam,  wAt, 64, 64, 3, 64);
  pack_w_kernel<<<ew(256LL * 576), 256, 0, stream>>>(w_cat, wAc, 256, 192, 3, 256);
  pack_w_kernel<<<ew(32LL * 64), 256, 0, stream>>>(pd_wq, wAq, 16, 64, 1, 32);
  pack_w_kernel<<<ew(64LL * 64), 256, 0, stream>>>(te_w, wAte, 64, 64, 1, 64);
  pack_w_kernel<<<ew(32LL * 64), 256, 0, stream>>>(ce_w, wAce, 4, 64, 1, 32);
  long long inTot = (long long)B * C * T * HW;
  cvt_in_kernel<<<ew(inTot), 256, 0, stream>>>(x1, xcat_bf, T, T2, 0, HW, inTot);
  cvt_in_kernel<<<ew(inTot), 256, 0, stream>>>(x2, xcat_bf, T, T2, T, HW, inTot);

  int slot = 0;
  auto nextStat = [&]() { return stats + 512 * (slot++); };

  auto conv = [&](float* out, const unsigned short* xb, const unsigned short* wA,
                  const float* bias, float* st, int Cout, int Mpad, int Cin, int KT,
                  int Tin, int Tout, int ts, int tp, int hw,
                  long long xbs, long long xcs, int awb) {
    dim3 grid((unsigned)((hw + 127) / 128), (unsigned)(B * Tout), (unsigned)(Mpad / 32));
    conv_wmma_kernel<<<grid, 128, 0, stream>>>(out, xb, wA, bias, st, Cout, Cin, KT,
                                               Cin * KT, Tin, Tout, ts, tp, hw, xbs, xcs, awb);
  };
  auto bnrelu = [&](const float* raw, const float* st, const float* g, const float* bb,
                    int Cout, long long nsp, float* f32dst, unsigned short* bf1,
                    unsigned short* bf2, int bfCtot, int bfC0) {
    long long total = (long long)B * Cout * nsp;
    float invcnt = 1.f / ((float)B * (float)nsp);
    bn_relu_kernel<<<ew(total), 256, 0, stream>>>(raw, st, g, bb, Cout, nsp, B, invcnt,
                                                  f32dst, bf1, bf2, bfCtot, bfC0);
  };

  // ---- conv1 on x1, x2, concat(x1,x2) ----
  long long xcs1 = (long long)T2 * HW;
  long long xbs1 = (long long)C * T2 * HW;
  { float* st = nextStat();
    conv(R, xcat_bf, wA1, nullptr, st, 64, 64, 256, 3, T, T, 1, 1, HW, xbs1, xcs1, 0);
    bnrelu(R, st, g_conv1, b_conv1, 64, Nsp, x1i_f, x1i_b, nullptr, 64, 0); }
  { float* st = nextStat();
    conv(R, xcat_bf + (long long)T * HW, wA1, nullptr, st, 64, 64, 256, 3, T, T, 1, 1, HW, xbs1, xcs1, 0);
    bnrelu(R, st, g_conv1, b_conv1, 64, Nsp, x2i_f, x2i_b, nullptr, 64, 0); }
  { float* st = nextStat();
    conv(R, xcat_bf, wA1, nullptr, st, 64, 64, 256, 3, T2, T2, 1, 1, HW, xbs1, xcs1, 0);
    bnrelu(R, st, g_conv1, b_conv1, 64, N2, xt1_f, xt1_b, nullptr, 64, 0); }

  // ---- PAM branches ----
  auto run_pam = [&](const float* xif, const unsigned short* xib, unsigned short* cat) {
    pam_pool_kernel<<<dim3(B * 64), 256, 0, stream>>>(xif, ppool, T, HW);
    pam_proj_kernel<<<dim3(4), 256, 0, stream>>>(ppool, pe_w, pe_g, pe_b, yenc, B);
    pam_kv_kernel<<<dim3(B), 256, 0, stream>>>(yenc, pd_wk, pd_bk, pd_wv, pd_bv, kbuf, vpad);
    conv(qbuf, xib, wAq, pd_bq, nullptr, 16, 32, 64, 1, 1, 1, 1, 0, Nsp, 64LL * Nsp, Nsp, 0);
    pam_attn_kernel<<<ew((long long)B * Nsp), 256, 0, stream>>>(qbuf, kbuf, attnP, Nsp, (long long)B * Nsp);
    conv(pamout, attnP, vpad, nullptr, nullptr, 64, 64, 64, 1, 1, 1, 1, 0, Nsp, 64LL * Nsp, Nsp, 64 * 64);
    axpy_bf_kernel<<<ew((long long)B * 64 * Nsp), 256, 0, stream>>>(pamout, xif, pd_scale, pres_b, (long long)B * 64 * Nsp);
    float* st = nextStat();
    conv(R, pres_b, wA2, nullptr, st, 64, 64, 64, 3, T, T, 1, 1, HW, 64LL * Nsp, Nsp, 0);
    bnrelu(R, st, g_conv2, b_conv2, 64, Nsp, nullptr, cat, nullptr, 192, 0);
  };
  run_pam(x1i_f, x1i_b, catA);
  run_pam(x2i_f, x2i_b, catB);

  // ---- CAM branches ----
  auto run_cam = [&](const float* xif, const unsigned short* xib, unsigned short* cat) {
    float* st = nextStat();
    conv(R, xib, wAce, nullptr, st, 4, 32, 64, 1, 1, 1, 1, 0, Nsp, 64LL * Nsp, Nsp, 0);
    bnrelu(R, st, ce_g, ce_b, 4, Nsp, yc, nullptr, nullptr, 4, 0);
    cam_energy_kernel<<<dim3(B * 64), 256, 0, stream>>>(xif, yc, eCam, Nsp);
    cam_apply_kernel<<<ew((long long)B * 64 * Nsp), 256, 0, stream>>>(xif, yc, eCam, cd_scale, cres_b, Nsp, (long long)B * 64 * Nsp);
    float* st2 = nextStat();
    conv(R, cres_b, wA2, nullptr, st2, 64, 64, 64, 3, T, T, 1, 1, HW, 64LL * Nsp, Nsp, 0);
    bnrelu(R, st2, g_conv2, b_conv2, 64, Nsp, nullptr, cat, nullptr, 192, 64);
  };
  run_cam(x1i_f, x1i_b, catB);   // x1_ca feeds out2's concat
  run_cam(x2i_f, x2i_b, catA);   // x2_ca feeds out1's concat

  // ---- TAM branch ----
  { float* st = nextStat();
    conv(R, xt1_b, wAte, nullptr, st, 64, 64, 64, 1, 1, 1, 1, 0, (int)N2, 64LL * N2, N2, 0);
    bnrelu(R, st, te_g, te_b, 64, N2, xt2_f, nullptr, nullptr, 64, 0);
    tam_energy_kernel<<<dim3(T2, T2, B), 256, 0, stream>>>(xt1_f, xt2_f, eTam, T2, HW);
    tam_attn_kernel<<<dim3(B), T2, 0, stream>>>(eTam, aTam, T2);
    tam_apply_kernel<<<ew((long long)B * 64 * N2), 256, 0, stream>>>(xt1_f, xt2_f, aTam, td_scale, tam_b, T2, HW, (long long)B * 64 * N2);
    float* st2 = nextStat();
    conv(R, tam_b, wAt, nullptr, st2, 64, 64, 64, 3, T2, T, 2, 1, HW, 64LL * N2, N2, 0);
    bnrelu(R, st2, g_tam, b_tam, 64, Nsp, nullptr, catA, catB, 192, 128); }

  // ---- final concat convs -> d_out ----
  float* out1 = (float*)d_out;
  float* out2 = out1 + (long long)B * 256 * Nsp;
  { float* st = nextStat();
    conv(R, catA, wAc, nullptr, st, 256, 256, 192, 3, T, T, 1, 1, HW, 192LL * Nsp, Nsp, 0);
    bnrelu(R, st, g_cat, b_cat, 256, Nsp, out1, nullptr, nullptr, 256, 0); }
  { float* st = nextStat();
    conv(R, catB, wAc, nullptr, st, 256, 256, 192, 3, T, T, 1, 1, HW, 192LL * Nsp, Nsp, 0);
    bnrelu(R, st, g_cat, b_cat, 256, Nsp, out2, nullptr, nullptr, 256, 0); }
}
